// GINWithJK_rec_38371237822823
// MI455X (gfx1250) — matrix operand order repaired and checked by reading, compiled, not run on hardware
//
#include <hip/hip_runtime.h>
#include <hip/hip_bf16.h>
#include <math.h>

// ---------------------------------------------------------------------------
// GINWithJK (+recurrent scorer) for MI455X (gfx1250, wave32, WMMA).
//
// Input buffer order assumption: top-level dict insertion order
//   0: x [N,64] f32   1: edge_index [2,E] i32   2: batch [N] i32
// then params flattened JAX-style (dict keys sorted alphabetically):
//   conv block (9): bn1.b bn1.g bn2.b bn2.g eps l1.b l1.w l2.b l2.w
//   3..11  conv1 ; 12..20 convs[0] ; 21..29 convs[1]
//   rec block (11): bn.b bn.g eps l1.b l1.w l2.b l2.w l3.b l3.w l4.b l4.w
//   30..40 convs_rec[0] ; 41..51 convs_rec[1]
//   52 lin1.b ; 53 lin1.w ; 54 lin2.b ; 55 lin2.w
// Output: [logits G*16 | x_rec N | em N*192]
// ---------------------------------------------------------------------------

#define BN_SCALE 0.9999950000374997f  // 1/sqrt(1+1e-5)

typedef __attribute__((ext_vector_type(16))) __bf16 v16bf;
typedef __attribute__((ext_vector_type(8)))  float  v8f;

union BF16Frag { v16bf v; unsigned u[8]; };

__device__ __forceinline__ unsigned short f2bf(float f) {
  unsigned u = __float_as_uint(f);
  unsigned r = u + 0x7fffu + ((u >> 16) & 1u);   // round-to-nearest-even
  return (unsigned short)(r >> 16);
}

// ---------------------------------------------------------------------------
__global__ void zero_kernel(float* __restrict__ p, long long n) {
  long long i = (long long)blockIdx.x * blockDim.x + threadIdx.x;
  long long stride = (long long)gridDim.x * blockDim.x;
  for (; i < n; i += stride) p[i] = 0.0f;
}

// segment_sum over edges: out[dst] += feat[src], 64 features, float4 granularity
__global__ void agg_kernel(const float* __restrict__ feat, int ldf,
                           float* __restrict__ out, int ldo,
                           const int* __restrict__ src, const int* __restrict__ dst,
                           int E) {
  int t = blockIdx.x * blockDim.x + threadIdx.x;
  int total = E * 16;
  if (t >= total) return;
  int e = t >> 4;
  int q = (t & 15) << 2;                          // column group 0..60 step 4
  int s = src[e];
  int d = dst[e];
  const float4 v = *(const float4*)(feat + (size_t)s * ldf + q);
  float* o = out + (size_t)d * ldo + q;
  atomicAdd(o + 0, v.x);
  atomicAdd(o + 1, v.y);
  atomicAdd(o + 2, v.z);
  atomicAdd(o + 3, v.w);
}

// ---------------------------------------------------------------------------
// Fused GEMM: Y[M,64] = act( (sA*A1 + A2)[M,K] @ W[K,64] + bias )
// mode 0: relu(z) ; mode 1: g*BNS*relu(z)+bb (conv) ; mode 2: relu(g*BNS*z+bb)
// Block = 256 threads (8 waves). Block tile = 32 rows x 64 cols.
// Each wave: one 16x16 tile via v_wmma_f32_16x16x32_bf16, K-loop step 32.
// ---------------------------------------------------------------------------
__global__ void gemm_bias_act(const float* __restrict__ A1, int lda1,
                              const float* __restrict__ A2, int lda2,
                              const float* __restrict__ epsP,
                              const float* __restrict__ W, int K,
                              const float* __restrict__ bias,
                              const float* __restrict__ bng,
                              const float* __restrict__ bnb,
                              int mode,
                              float* __restrict__ out, int ldo, int M) {
  __shared__ __align__(16) unsigned short aLds[32 * 194];   // 32 x (K+2), K<=192
  __shared__ __align__(16) unsigned short wLds[64 * 194];   // 64 x (K+2), [col][k]

  const int tid  = threadIdx.x;
  const int lane = tid & 31;
  const int wave = tid >> 5;
  const int AK = K + 2;
  const int WK = K + 2;
  const int rowBase = blockIdx.x * 32;
  const float sA = 1.0f + (epsP ? epsP[0] : 0.0f);

  // Stage weights (fp32 -> bf16), transposed to [col][k] for contiguous B frags
  for (int i = tid; i < K * 64; i += 256) {
    int k = i >> 6;
    int n = i & 63;
    wLds[n * WK + k] = f2bf(W[i]);
  }
  // Stage A rows with fused (1+eps)*h + agg, fp32 -> bf16
  for (int i = tid; i < 32 * K; i += 256) {
    int r = i / K;
    int k = i - r * K;
    int row = rowBase + r;
    float v = 0.0f;
    if (row < M) {
      v = sA * A1[(size_t)row * lda1 + k];
      if (A2) v += A2[(size_t)row * lda2 + k];
    }
    aLds[r * AK + k] = f2bf(v);
  }
  __syncthreads();

  const int r0 = (wave >> 2) * 16;        // row tile: 0 or 16
  const int n0 = (wave & 3) * 16;         // col tile: 0,16,32,48
  const int m  = lane & 15;
  const int nl = lane & 15;
  const int khalf = lane >> 4;

  v8f acc = {};
  for (int kk = 0; kk < K; kk += 32) {
    BF16Frag a, b;
    // A 16x32 bf16 fragment: element e -> k = kk + (e>>3)*16 + khalf*8 + (e&7)
    const unsigned short* arow = &aLds[(r0 + m) * AK];
#pragma unroll
    for (int e2 = 0; e2 < 8; ++e2) {
      int e = e2 * 2;
      int k = kk + ((e >> 3) << 4) + khalf * 8 + (e & 7);
      a.u[e2] = *(const unsigned*)&arow[k];
    }
    // B 32x16 bf16 fragment: element e -> k = kk + (e>>3)*16 + khalf*8 + (e&7), col = n0+nl
    const unsigned short* bcol = &wLds[(n0 + nl) * WK + kk + khalf * 8];
#pragma unroll
    for (int e2 = 0; e2 < 8; ++e2) {
      b.u[e2] = *(const unsigned*)&bcol[((e2 >> 2) << 4) + (e2 & 3) * 2];
    }
    acc = __builtin_amdgcn_wmma_f32_16x16x32_bf16(
        false, a.v, false, b.v, (short)0, acc, false, false);
  }

  // Epilogue: C/D layout lane l -> col = n0 + (l&15), rows m = (l>>4)*8 + r
  const int col = n0 + nl;
  const float bv = bias[col];
  float gg = 0.0f, bb = 0.0f;
  if (mode != 0) { gg = bng[col] * BN_SCALE; bb = bnb[col]; }
#pragma unroll
  for (int r = 0; r < 8; ++r) {
    int row = rowBase + r0 + khalf * 8 + r;
    if (row < M) {
      float z = acc[r] + bv;
      float y;
      if (mode == 0)      y = fmaxf(z, 0.0f);
      else if (mode == 1) y = gg * fmaxf(z, 0.0f) + bb;
      else                y = fmaxf(gg * z + bb, 0.0f);
      out[(size_t)row * ldo + col] = y;
    }
  }
}

// x_rec = H[M,64] @ w4[64] + b4 : lane pair per row (32 floats each half)
__global__ void rowdot_kernel(const float* __restrict__ H,
                              const float* __restrict__ w4,
                              const float* __restrict__ b4,
                              float* __restrict__ out, int M) {
  int t = blockIdx.x * blockDim.x + threadIdx.x;
  int lane = t & 31;
  int row = (t >> 5) * 16 + (lane >> 1);
  int half = lane & 1;
  float s = 0.0f;
  if (row < M) {
    const float* h = H + (size_t)row * 64 + half * 32;
    const float* w = w4 + half * 32;
#pragma unroll
    for (int j = 0; j < 8; ++j) {
      float4 a = *(const float4*)(h + j * 4);
      float4 b = *(const float4*)(w + j * 4);
      s += a.x * b.x + a.y * b.y + a.z * b.z + a.w * b.w;
    }
  }
  s += __shfl_xor(s, 1, 32);
  if (half == 0 && row < M) out[row] = s + b4[0];
}

// mean-pool scatter: pooled[batch[n]] += em[n] (192 cols, float4 granularity)
__global__ void pool_kernel(const float* __restrict__ em,
                            const int* __restrict__ batch,
                            float* __restrict__ pooled, int N) {
  int t = blockIdx.x * blockDim.x + threadIdx.x;
  int total = N * 48;
  if (t >= total) return;
  int node = t / 48;
  int q = (t - node * 48) << 2;
  int b = batch[node];
  const float4 v = *(const float4*)(em + (size_t)node * 192 + q);
  float* o = pooled + (size_t)b * 192 + q;
  atomicAdd(o + 0, v.x);
  atomicAdd(o + 1, v.y);
  atomicAdd(o + 2, v.z);
  atomicAdd(o + 3, v.w);
}

__global__ void count_kernel(const int* __restrict__ batch,
                             float* __restrict__ counts, int N) {
  int t = blockIdx.x * blockDim.x + threadIdx.x;
  if (t < N) atomicAdd(&counts[batch[t]], 1.0f);
}

__global__ void divide_kernel(float* __restrict__ pooled,
                              const float* __restrict__ counts, int G) {
  int t = blockIdx.x * blockDim.x + threadIdx.x;
  if (t >= G * 192) return;
  pooled[t] *= 1.0f / fmaxf(counts[t / 192], 1.0f);
}

// logits = H[G,64] @ W2[64,16] + b2 ; log_softmax over 16 classes
__global__ void head_kernel(const float* __restrict__ H,
                            const float* __restrict__ W2,
                            const float* __restrict__ b2,
                            float* __restrict__ out, int G) {
  int g = blockIdx.x;
  if (g >= G) return;
  int lane = threadIdx.x;
  int n = lane & 15;                        // lanes 16-31 mirror lanes 0-15
  float z = 0.0f;
  const float* h = H + (size_t)g * 64;
#pragma unroll 8
  for (int k = 0; k < 64; ++k) z += h[k] * W2[k * 16 + n];
  z += b2[n];
  float mx = z;
  for (int d = 1; d < 16; d <<= 1) mx = fmaxf(mx, __shfl_xor(mx, d, 16));
  float ex = __expf(z - mx);
  float sm = ex;
  for (int d = 1; d < 16; d <<= 1) sm += __shfl_xor(sm, d, 16);
  float lsm = z - mx - __logf(sm);
  if (lane < 16) out[(size_t)g * 16 + n] = lsm;
}

// ---------------------------------------------------------------------------
extern "C" void kernel_launch(void* const* d_in, const int* in_sizes, int n_in,
                              void* d_out, int out_size, void* d_ws, size_t ws_size,
                              hipStream_t stream) {
  const float* x   = (const float*)d_in[0];
  const int* ei    = (const int*)d_in[1];
  const int* batch = (const int*)d_in[2];
  const int N = in_sizes[2];
  const int E = in_sizes[1] / 2;
  const int G = (out_size - N - N * 192) / 16;
  const int* src = ei;
  const int* dst = ei + E;

  auto F = [&](int i) { return (const float*)d_in[i]; };

  float* outLogits = (float*)d_out;
  float* outXrec   = outLogits + (size_t)G * 16;
  float* outEm     = outXrec + N;

  float* ws      = (float*)d_ws;
  float* agg0    = ws;                                // [N,64]
  float* aggbuf  = agg0 + (size_t)N * 64;             // [N,192] = [agg(h1)|agg(h2)|agg(h3)]
  float* tmpA    = aggbuf + (size_t)N * 192;          // [N,64]
  float* tmpB    = tmpA + (size_t)N * 64;             // [N,64]
  float* pooled  = tmpB + (size_t)N * 64;             // [G,192]
  float* counts  = pooled + (size_t)G * 192;          // [G]

  const int gridN = (N + 31) / 32;
  const int gridG = (G + 31) / 32;
  const int aggBlocks = (E * 16 + 255) / 256;

  // zero accumulation buffers (agg0+aggbuf contiguous; pooled+counts contiguous)
  zero_kernel<<<1024, 256, 0, stream>>>(agg0, (long long)N * 256);
  zero_kernel<<<64, 256, 0, stream>>>(pooled, (long long)G * 192 + G);

  // conv MLP helper: two fused GEMMs (Linear-ReLU-BN twice)
  auto conv = [&](const float* hin, int ldh, const float* agg, int ldagg,
                  int b, float* hout, int ldo) {
    gemm_bias_act<<<gridN, 256, 0, stream>>>(
        hin, ldh, agg, ldagg, F(b + 4), F(b + 6), 64, F(b + 5),
        F(b + 1), F(b + 0), 1, tmpA, 64, N);
    gemm_bias_act<<<gridN, 256, 0, stream>>>(
        tmpA, 64, nullptr, 0, nullptr, F(b + 8), 64, F(b + 7),
        F(b + 3), F(b + 2), 1, hout, ldo, N);
  };

  // conv1 : agg(x) -> h1 (em cols 0..63)
  agg_kernel<<<aggBlocks, 256, 0, stream>>>(x, 64, agg0, 64, src, dst, E);
  conv(x, 64, agg0, 64, 3, outEm + 0, 192);
  // conv2 : agg(h1) (also reused by rec layer) -> h2 (em cols 64..127)
  agg_kernel<<<aggBlocks, 256, 0, stream>>>(outEm + 0, 192, aggbuf + 0, 192, src, dst, E);
  conv(outEm + 0, 192, aggbuf + 0, 192, 12, outEm + 64, 192);
  // conv3 : agg(h2) -> h3 (em cols 128..191)
  agg_kernel<<<aggBlocks, 256, 0, stream>>>(outEm + 64, 192, aggbuf + 64, 192, src, dst, E);
  conv(outEm + 64, 192, aggbuf + 64, 192, 21, outEm + 128, 192);
  // agg(h3) completes agg(em) = aggbuf
  agg_kernel<<<aggBlocks, 256, 0, stream>>>(outEm + 128, 192, aggbuf + 128, 192, src, dst, E);

  // recurrent scorer: only the LAST convs_rec iteration survives (em is constant)
  const int rb = 41;
  gemm_bias_act<<<gridN, 256, 0, stream>>>(
      outEm, 192, aggbuf, 192, F(rb + 2), F(rb + 4), 192, F(rb + 3),
      nullptr, nullptr, 0, tmpA, 64, N);
  gemm_bias_act<<<gridN, 256, 0, stream>>>(
      tmpA, 64, nullptr, 0, nullptr, F(rb + 6), 64, F(rb + 5),
      nullptr, nullptr, 0, tmpB, 64, N);
  gemm_bias_act<<<gridN, 256, 0, stream>>>(
      tmpB, 64, nullptr, 0, nullptr, F(rb + 8), 64, F(rb + 7),
      F(rb + 1), F(rb + 0), 2, tmpA, 64, N);
  rowdot_kernel<<<((N + 15) / 16 * 32 + 255) / 256, 256, 0, stream>>>(
      tmpA, F(rb + 10), F(rb + 9), outXrec, N);

  // graph mean-pool + classification head
  pool_kernel<<<(N * 48 + 255) / 256, 256, 0, stream>>>(outEm, batch, pooled, N);
  count_kernel<<<(N + 255) / 256, 256, 0, stream>>>(batch, counts, N);
  divide_kernel<<<(G * 192 + 255) / 256, 256, 0, stream>>>(pooled, counts, G);
  gemm_bias_act<<<gridG, 256, 0, stream>>>(
      pooled, 192, nullptr, 0, nullptr, F(53), 192, F(52),
      nullptr, nullptr, 0, tmpB, 64, G);
  head_kernel<<<G, 32, 0, stream>>>(tmpB, F(55), F(54), outLogits, G);
}
